// MultiHeadedAttention_47175920779651
// MI455X (gfx1250) — matrix operand order repaired and checked by reading, not compiled, not run on hardware
//
#include <hip/hip_runtime.h>
#include <hip/hip_bf16.h>
#include <cmath>

// MI455X / gfx1250, wave32. Fused attention using v_wmma_f32_16x16x32_f16,
// GLOBAL_LOAD_ASYNC_TO_LDS_B128 (mask tile), NT cache hints on touch-once
// streams (prob/out stores, rel/ts loads) so q/k/vT stay resident in L2.
//
// Pipeline:
//   proj_wmma  x3 : q16(b,h,s,hd), k16(b,h,s,hd), vT16(b,h,hd,s)  [f16, in d_ws]
//   attn_wmma  x1 : per (b, 16-row q-tile) workgroup:
//       async-copy mask tile -> LDS
//       phase1: C = (1-l1)*l2*softmax(time) + l1*softmax(rel) into LDS (head-indep)
//       per head: WMMA scores -> LDS, row softmax, blend+write prob (NT, float4),
//                 prob(f16) -> LDS, WMMA prob@v -> out (NT)
//
// Workspace use: 3 * 8*1024*1024 f16 = 50.3 MB.

typedef __attribute__((ext_vector_type(16))) _Float16 v16h;
typedef __attribute__((ext_vector_type(8)))  _Float16 v8h;
typedef __attribute__((ext_vector_type(4)))  _Float16 v4h;
typedef __attribute__((ext_vector_type(8)))  float    v8f;
typedef __attribute__((ext_vector_type(4)))  float    v4f;

union F16x16 { v16h v; v8h h[2]; _Float16 e[16]; };

// ---------------------------------------------------------------------------
// Projection: Y = X @ W^T + bias, X:(8192,1024) f32, W:(1024,1024) f32.
// mode 0: out[b,h,s,hd] (f16)   mode 1: out[b,h,hd,s] (f16, transposed for V)
// grid (256,16), block 256 (8 waves). Wave w: M-subtile w>>2, N-subtile w&3.
// ---------------------------------------------------------------------------
__global__ __launch_bounds__(256)
void proj_wmma(const float* __restrict__ X, const float* __restrict__ W,
               const float* __restrict__ bias, _Float16* __restrict__ out, int mode)
{
    const int lane = threadIdx.x & 31;
    const int w    = threadIdx.x >> 5;
    const int half = lane >> 4;        // 0 | 1
    const int l16  = lane & 15;
    const int m_base = blockIdx.x * 32 + (w >> 2) * 16;
    const int n_base = blockIdx.y * 64 + (w & 3) * 16;

    const size_t arow = (size_t)(m_base + l16) * 1024;   // A: row = token
    const size_t brow = (size_t)(n_base + l16) * 1024;   // B: row n of W (y=x@W^T)
    const int g = half * 8;

    v8f acc = {};
    for (int kc = 0; kc < 1024; kc += 32) {
        // A-fragment (16x32 f16): lane holds K = {kc+g..+7, kc+16+g..+7}
        v8f x0 = *(const v8f*)(X + arow + kc + g);
        v8f x1 = *(const v8f*)(X + arow + kc + g + 16);
        // B-fragment (32x16 f16): lane n holds K = {kc+g..+7, kc+16+g..+7}
        v8f w0 = *(const v8f*)(W + brow + kc + g);
        v8f w1 = *(const v8f*)(W + brow + kc + g + 16);
        F16x16 a, b;
#pragma unroll
        for (int i = 0; i < 8; ++i) {
            a.e[i] = (_Float16)x0[i]; a.e[8 + i] = (_Float16)x1[i];
            b.e[i] = (_Float16)w0[i]; b.e[8 + i] = (_Float16)w1[i];
        }
        acc = __builtin_amdgcn_wmma_f32_16x16x32_f16(false, a.v, false, b.v,
                                                     (short)0, acc, false, false);
    }
#pragma unroll
    for (int r = 0; r < 8; ++r) {
        const int m  = r + half * 8;           // C/D layout
        const int gm = m_base + m;
        const int gn = n_base + l16;
        const float val = acc[r] + bias[gn];
        const int bidx = gm >> 10, s = gm & 1023;
        const int hh = gn >> 6,    hd = gn & 63;
        if (mode == 0)
            out[(((size_t)bidx * 16 + hh) * 1024 + s) * 64 + hd] = (_Float16)val;
        else
            out[(((size_t)bidx * 16 + hh) * 64 + hd) * 1024 + s] = (_Float16)val;
    }
}

// ---------------------------------------------------------------------------
// Fused attention. grid 512 = B * (S/16), block 256 (8 waves).
// LDS: sC 64KB | sS 64KB | sP 32KB | sM 16KB  = 176 KB dynamic.
// ---------------------------------------------------------------------------
__global__ __launch_bounds__(256)
void attn_wmma(const _Float16* __restrict__ q16, const _Float16* __restrict__ k16,
               const _Float16* __restrict__ vT16,
               const float* __restrict__ rel, const float* __restrict__ ts,
               const unsigned char* __restrict__ mask,
               const float* __restrict__ l1p, const float* __restrict__ l2p,
               float* __restrict__ out, float* __restrict__ prob)
{
    extern __shared__ char smem[];
    float*         sC = (float*)smem;                    // 16x1024 combined corr.
    float*         sS = (float*)(smem + 65536);          // 16x1024 scores/scratch
    _Float16*      sP = (_Float16*)(smem + 131072);      // 16x1024 prob f16
    unsigned char* sM = (unsigned char*)(smem + 163840); // 16x1024 mask bytes

    const int b      = blockIdx.x >> 6;
    const int q_base = (blockIdx.x & 63) * 16;
    const int lane = threadIdx.x & 31;
    const int w    = threadIdx.x >> 5;
    const int half = lane >> 4;
    const int l16  = lane & 15;

    // ---- Async copy of the contiguous 16 KB mask tile into sM.
    //      GLOBAL_LOAD_ASYNC_TO_LDS_B128: dst VGPR = LDS byte offset
    //      (low 32 bits of generic LDS address), addr = 64-bit VGPR pair;
    //      inst offset applies to both sides (ISA 08_async_tensor §4.4).
    {
        const unsigned char* gsrc =
            mask + ((size_t)b * 1024 + q_base) * 1024 + (size_t)threadIdx.x * 16;
        unsigned lds_off = (unsigned)(size_t)sM + threadIdx.x * 16u;
        unsigned long long ga = (unsigned long long)(size_t)gsrc;
        asm volatile(
            "global_load_async_to_lds_b128 %0, %1, off\n\t"
            "global_load_async_to_lds_b128 %0, %1, off offset:4096\n\t"
            "global_load_async_to_lds_b128 %0, %1, off offset:8192\n\t"
            "global_load_async_to_lds_b128 %0, %1, off offset:12288"
            :: "v"(lds_off), "v"(ga) : "memory");
        asm volatile("s_wait_asynccnt 0x0" ::: "memory");
    }
    __syncthreads();

    const float l1 = l1p[0], l2 = l2p[0];
    const float c1 = l1;                           // weight of rel_attn
    const float c2 = (1.0f - l1) * l2;             // weight of time_attn
    const float alpha = (1.0f - l1) * (1.0f - l2); // weight of score softmax

    // ---- Phase 1: head-independent correction term C into sC.
    //      One row per wave iteration; 32 lanes x float4 = 512B/wave-instr.
    //      rel/ts are touch-once: NT loads keep them out of L2.
    for (int rr = 0; rr < 2; ++rr) {
        const int row = 2 * w + rr;                         // 0..15
        const size_t gb = ((size_t)b * 1024 + (q_base + row)) * 1024;
        float tmax = -3.4e38f, rmax = -3.4e38f;
#pragma unroll 2
        for (int j = 0; j < 8; ++j) {
            const int c0 = (j * 32 + lane) * 4;
            const unsigned mku = *(const unsigned*)(sM + row * 1024 + c0);
            const v4f t4 = __builtin_nontemporal_load((const v4f*)(ts  + gb + c0));
            const v4f r4 = __builtin_nontemporal_load((const v4f*)(rel + gb + c0));
            v4f to, ro;
#pragma unroll
            for (int i = 0; i < 4; ++i) {
                const bool mm = (mku >> (8 * i)) & 0xff;
                // time_stamp = where(m, -1e5, exp(-|ts|))
                to[i] = mm ? -100000.0f : __expf(-fabsf(t4[i]));
                // rel_m = rel*mask_f; where(rel_m==0, -1e5, rel_m)
                ro[i] = (mm && r4[i] != 0.0f) ? r4[i] : -100000.0f;
                tmax = fmaxf(tmax, to[i]); rmax = fmaxf(rmax, ro[i]);
            }
            *(v4f*)(sS + row * 1024 + c0) = to;
            *(v4f*)(sC + row * 1024 + c0) = ro;
        }
        for (int off = 16; off; off >>= 1) {
            tmax = fmaxf(tmax, __shfl_xor(tmax, off, 32));
            rmax = fmaxf(rmax, __shfl_xor(rmax, off, 32));
        }
        float tsum = 0.f, rsum = 0.f;
#pragma unroll 2
        for (int j = 0; j < 8; ++j) {
            const int c0 = (j * 32 + lane) * 4;
            const v4f t4 = *(const v4f*)(sS + row * 1024 + c0);
            const v4f r4 = *(const v4f*)(sC + row * 1024 + c0);
#pragma unroll
            for (int i = 0; i < 4; ++i) {
                tsum += __expf(t4[i] - tmax);
                rsum += __expf(r4[i] - rmax);
            }
        }
        for (int off = 16; off; off >>= 1) {
            tsum += __shfl_xor(tsum, off, 32);
            rsum += __shfl_xor(rsum, off, 32);
        }
        const float tinv = 1.0f / tsum, rinv = 1.0f / rsum;
#pragma unroll 2
        for (int j = 0; j < 8; ++j) {
            const int c0 = (j * 32 + lane) * 4;
            const v4f t4 = *(const v4f*)(sS + row * 1024 + c0);
            const v4f r4 = *(const v4f*)(sC + row * 1024 + c0);
            v4f c;
#pragma unroll
            for (int i = 0; i < 4; ++i)
                c[i] = c2 * __expf(t4[i] - tmax) * tinv
                     + c1 * __expf(r4[i] - rmax) * rinv;
            *(v4f*)(sC + row * 1024 + c0) = c;
        }
    }
    __syncthreads();

    // ---- Per-head: scores -> softmax -> blend/write prob -> prob@v -> out
    for (int h = 0; h < 16; ++h) {
        const size_t bh = (size_t)(b * 16 + h);

        // Prefetch next head's K / vT rows (global_prefetch_b8 path).
        if (h + 1 < 16) {
            const size_t nbh = bh + 1;
            __builtin_prefetch(k16  + (nbh * 1024 + (int)threadIdx.x * 4) * 64, 0, 1);
            __builtin_prefetch(vT16 + (nbh * 64 + (threadIdx.x >> 4)) * 1024
                                    + (threadIdx.x & 15) * 64, 0, 1);
        }

        // (a) scores: wave w covers keys [w*128, w*128+128). A-frags hoisted.
        {
            const _Float16* qp = q16 + (bh * 1024 + q_base + l16) * 64;
            F16x16 a0, a1;                     // kc = 0 and kc = 32
            a0.h[0] = *(const v8h*)(qp + half * 16);
            a0.h[1] = *(const v8h*)(qp + half * 16 + 8);
            a1.h[0] = *(const v8h*)(qp + 32 + half * 16);
            a1.h[1] = *(const v8h*)(qp + 32 + half * 16 + 8);
#pragma unroll 4
            for (int kt = 0; kt < 8; ++kt) {
                const int key_base = w * 128 + kt * 16;
                const _Float16* kp = k16 + (bh * 1024 + key_base + l16) * 64;
                const v16h b0 = *(const v16h*)(kp + half * 16);
                const v16h b1 = *(const v16h*)(kp + 32 + half * 16);
                v8f acc = {};
                acc = __builtin_amdgcn_wmma_f32_16x16x32_f16(
                    false, a0.v, false, b0, (short)0, acc, false, false);
                acc = __builtin_amdgcn_wmma_f32_16x16x32_f16(
                    false, a1.v, false, b1, (short)0, acc, false, false);
#pragma unroll
                for (int r = 0; r < 8; ++r) {
                    const int m = r + half * 8;
                    const int col = key_base + l16;
                    const unsigned char mk = sM[m * 1024 + col];
                    sS[m * 1024 + col] = mk ? -1e32f : acc[r] * 0.125f;
                }
            }
        }
        __syncthreads();

        // (b) row softmax + blend + prob (NT float4 stores) + sP(f16)
        for (int rr = 0; rr < 2; ++rr) {
            const int row = 2 * w + rr;
            float smax = -3.4e38f;
#pragma unroll 2
            for (int j = 0; j < 8; ++j) {
                const v4f s4 = *(const v4f*)(sS + row * 1024 + (j * 32 + lane) * 4);
                smax = fmaxf(smax, fmaxf(fmaxf(s4[0], s4[1]), fmaxf(s4[2], s4[3])));
            }
            for (int off = 16; off; off >>= 1)
                smax = fmaxf(smax, __shfl_xor(smax, off, 32));
            float ssum = 0.f;
#pragma unroll 2
            for (int j = 0; j < 8; ++j) {
                const v4f s4 = *(const v4f*)(sS + row * 1024 + (j * 32 + lane) * 4);
#pragma unroll
                for (int i = 0; i < 4; ++i) ssum += __expf(s4[i] - smax);
            }
            for (int off = 16; off; off >>= 1)
                ssum += __shfl_xor(ssum, off, 32);
            const float sinv = 1.0f / ssum;
            float* pg = prob + (bh * 1024 + q_base + row) * 1024;
#pragma unroll 2
            for (int j = 0; j < 8; ++j) {
                const int c0 = (j * 32 + lane) * 4;
                const v4f s4 = *(const v4f*)(sS + row * 1024 + c0);
                const v4f c4 = *(const v4f*)(sC + row * 1024 + c0);
                v4f p;
#pragma unroll
                for (int i = 0; i < 4; ++i)
                    p[i] = alpha * __expf(s4[i] - smax) * sinv + c4[i];
                __builtin_nontemporal_store(p, (v4f*)(pg + c0)); // 512B/wave, NT
                v4h ph = { (_Float16)p[0], (_Float16)p[1],
                           (_Float16)p[2], (_Float16)p[3] };
                *(v4h*)(sP + row * 1024 + c0) = ph;
            }
        }
        __syncthreads();

        // (c) out tile (16x64) = sP(16x1024) @ v(1024x64); waves split N x K/2
        {
            const int ntile = w & 3, khalf = w >> 2;
            const int n_base = ntile * 16;
            const _Float16* vp = vT16 + (bh * 64 + n_base + l16) * 1024;
            const _Float16* pp = sP + l16 * 1024;
            v8f acc = {};
#pragma unroll 4
            for (int kk = khalf * 512; kk < khalf * 512 + 512; kk += 32) {
                F16x16 a;
                a.h[0] = *(const v8h*)(pp + kk + half * 16);
                a.h[1] = *(const v8h*)(pp + kk + half * 16 + 8);
                const v16h bm = *(const v16h*)(vp + kk + half * 16);
                acc = __builtin_amdgcn_wmma_f32_16x16x32_f16(
                    false, a.v, false, bm, (short)0, acc, false, false);
            }
            if (khalf == 1) {
#pragma unroll
                for (int r = 0; r < 8; ++r) {
                    const int m = r + half * 8;
                    sS[ntile * 256 + m * 16 + l16] = acc[r];   // scratch (scores dead)
                }
            }
            __syncthreads();
            if (khalf == 0) {
#pragma unroll
                for (int r = 0; r < 8; ++r) {
                    const int m = r + half * 8;
                    const float val = acc[r] + sS[ntile * 256 + m * 16 + l16];
                    __builtin_nontemporal_store(
                        val, out + ((size_t)b * 1024 + q_base + m) * 1024
                                 + h * 64 + n_base + l16);
                }
            }
        }
        __syncthreads();
    }
}

// ---------------------------------------------------------------------------
extern "C" void kernel_launch(void* const* d_in, const int* in_sizes, int n_in,
                              void* d_out, int out_size, void* d_ws, size_t ws_size,
                              hipStream_t stream)
{
    (void)in_sizes; (void)n_in; (void)out_size; (void)ws_size;

    const float* query = (const float*)d_in[0];
    const float* key_  = (const float*)d_in[1];
    const float* value = (const float*)d_in[2];
    const float* rel   = (const float*)d_in[3];
    const float* ts    = (const float*)d_in[4];
    const unsigned char* mask = (const unsigned char*)d_in[5]; // jnp bool -> 1 byte
    const float* l1 = (const float*)d_in[6];
    const float* l2 = (const float*)d_in[7];
    // d_in[8] = encode_pos (unused)
    const float* Wq = (const float*)d_in[9];
    const float* bq = (const float*)d_in[10];
    const float* Wk = (const float*)d_in[11];
    const float* bk = (const float*)d_in[12];
    const float* Wv = (const float*)d_in[13];
    const float* bv = (const float*)d_in[14];

    _Float16* q16  = (_Float16*)d_ws;                       // (b,h,s,hd)
    _Float16* k16  = q16 + (size_t)8 * 1024 * 1024;         // (b,h,s,hd)
    _Float16* vT16 = k16 + (size_t)8 * 1024 * 1024;         // (b,h,hd,s)

    float* out  = (float*)d_out;                            // (B,S,D)
    float* prob = out + (size_t)8 * 1024 * 1024;            // (B,H,S,S)

    dim3 pgrid(256, 16), pblock(256);
    proj_wmma<<<pgrid, pblock, 0, stream>>>(query, Wq, bq, q16, 0);
    proj_wmma<<<pgrid, pblock, 0, stream>>>(key_, Wk, bk, k16, 0);
    proj_wmma<<<pgrid, pblock, 0, stream>>>(value, Wv, bv, vT16, 1);

    const size_t smem = 65536 + 65536 + 32768 + 16384;      // 180224 B
    hipFuncSetAttribute(reinterpret_cast<const void*>(attn_wmma),
                        hipFuncAttributeMaxDynamicSharedMemorySize, (int)smem);
    attn_wmma<<<dim3(512), dim3(256), smem, stream>>>(q16, k16, vT16, rel, ts,
                                                      mask, l1, l2, out, prob);
}
